// MultiQueryAttention_68118181315078
// MI455X (gfx1250) — matrix operand order repaired and checked by reading, compiled, not run on hardware
//
#include <hip/hip_runtime.h>

#define USE_ASYNC 1

typedef __attribute__((ext_vector_type(8)))  float   v8f;
typedef __attribute__((ext_vector_type(8)))  __bf16  v8bf;
typedef __attribute__((ext_vector_type(16))) __bf16  v16bf;

__device__ __forceinline__ __bf16 f32_to_bf16(float f) {
  unsigned u = __builtin_bit_cast(unsigned, f);
  u += 0x7FFFu + ((u >> 16) & 1u);              // round-to-nearest-even
  unsigned short h = (unsigned short)(u >> 16);
  return __builtin_bit_cast(__bf16, h);
}

__device__ __forceinline__ v16bf cat8(v8bf lo, v8bf hi) {
  v16bf r;
#pragma unroll
  for (int i = 0; i < 8; ++i) { r[i] = lo[i]; r[i + 8] = hi[i]; }
  return r;
}

// A-matrix 16x32 bf16: lane holds row (lane&15), 16 contiguous K at (lane>>4)*16.
// Caller passes p = rowbase + (lane>>4)*16; second chunk at +8 elements.
__device__ __forceinline__ v16bf load_a_frag(const __bf16* p) {
  return cat8(*(const v8bf*)p, *(const v8bf*)(p + 8));
}
// B-matrix 32x16: lane holds col (lane&15), 16 contiguous K at (lane>>4)*16.
// Caller passes fully-resolved pointer; two 16B halves (16B alignment only).
__device__ __forceinline__ v16bf load_b_frag(const __bf16* p) {
  return cat8(*(const v8bf*)p, *(const v8bf*)(p + 8));
}

__device__ __forceinline__ v8f wmma_bf16(v16bf a, v16bf b, v8f c) {
  return __builtin_amdgcn_wmma_f32_16x16x32_bf16(false, a, false, b, (short)0, c, false, false);
}

// 32 contiguous bytes global -> LDS (per-thread), async on CDNA5.
__device__ __forceinline__ void stage32(__bf16* lds, const __bf16* g) {
#if USE_ASYNC
  unsigned l = (unsigned)(uintptr_t)lds;
  unsigned long long ga = (unsigned long long)(uintptr_t)g;
  asm volatile("global_load_async_to_lds_b128 %0, %1, off"
               :: "v"(l), "v"(ga) : "memory");
  asm volatile("global_load_async_to_lds_b128 %0, %1, off"
               :: "v"(l + 16u), "v"(ga + 16ull) : "memory");
#else
  *(v8bf*)lds       = *(const v8bf*)g;
  *(v8bf*)(lds + 8) = *(const v8bf*)(g + 8);
#endif
}
// wait for ALL outstanding async transfers
__device__ __forceinline__ void stage_wait0() {
#if USE_ASYNC
  asm volatile("s_wait_asynccnt 0x0" ::: "memory");
#endif
}
// wait until only the newest 4 transfers (next block, 4 per wave) remain
__device__ __forceinline__ void stage_wait4() {
#if USE_ASYNC
  asm volatile("s_wait_asynccnt 0x4" ::: "memory");
#endif
}

// ---------------------------------------------------------------- conversions
__global__ void __launch_bounds__(256) cvt_bf16_kernel(const float* __restrict__ s,
                                                       __bf16* __restrict__ d, int n) {
  int i = blockIdx.x * 256 + threadIdx.x;
  if (i < n) d[i] = f32_to_bf16(s[i]);
}

// v [B,L,128] f32 -> vt [B,128,L] bf16 (so P@V B-fragments are contiguous-K)
__global__ void __launch_bounds__(256) vtrans_kernel(const float* __restrict__ v,
                                                     __bf16* __restrict__ vt,
                                                     int B, int L) {
  int i = blockIdx.x * 256 + threadIdx.x;
  if (i >= B * L * 128) return;
  int d = i & 127;
  int l = (i >> 7) % L;
  int b = i / (L * 128);
  vt[((size_t)(b * 128 + d)) * L + l] = f32_to_bf16(v[i]);
}

// ------------------------------------------------- fused RoPE + RMSNorm (bf16 out)
// in : [B, L, H*128] f32 rows (head h at column h*128)   (H=1 for the K path)
// out: [B, H, L, 128] bf16
__global__ void __launch_bounds__(256) rope_rms_kernel(
    const float* __restrict__ xin, const float* __restrict__ cosT,
    const float* __restrict__ sinT, __bf16* __restrict__ xout,
    int B, int H, int L) {
  int wid  = blockIdx.x * 8 + (threadIdx.x >> 5);
  int lane = threadIdx.x & 31;
  if (wid >= B * H * L) return;
  int l = wid % L;
  int h = (wid / L) % H;
  int b = wid / (L * H);
  const float* in  = xin + ((size_t)(b * L + l)) * (H * 128) + h * 128;
  __bf16*      out = xout + ((size_t)((b * H + h) * L + l)) * 128;
  const float* cr = cosT + (size_t)l * 64;
  const float* sr = sinT + (size_t)l * 64;
  int i0 = lane, i1 = lane + 32;
  float c0 = cr[i0], s0 = sr[i0], c1 = cr[i1], s1 = sr[i1];
  float a0 = in[i0], b0 = in[i0 + 64], a1 = in[i1], b1 = in[i1 + 64];
  float y0 =  a0 * c0 + b0 * s0;
  float y1 = -a0 * s0 + b0 * c0;
  float y2 =  a1 * c1 + b1 * s1;
  float y3 = -a1 * s1 + b1 * c1;
  float ssq = y0 * y0 + y1 * y1 + y2 * y2 + y3 * y3;
#pragma unroll
  for (int m = 1; m < 32; m <<= 1) ssq += __shfl_xor(ssq, m, 32);
  float r = rsqrtf(ssq * (1.0f / 128.0f) + 1.1920929e-07f);
  out[i0]      = f32_to_bf16(y0 * r);
  out[i0 + 64] = f32_to_bf16(y1 * r);
  out[i1]      = f32_to_bf16(y2 * r);
  out[i1 + 64] = f32_to_bf16(y3 * r);
}

// ------------------------------------------------------------------- GEMM (NT)
// C[M,N] f32 = A[M,K]bf16 * W[N,K]bf16^T.  One wave computes a 64x64 C tile
// (16 WMMAs / K-step, 32 FLOP per L2 byte; operands live in 192MB L2).
__global__ void __launch_bounds__(256) gemm_bf16_nt(
    const __bf16* __restrict__ A, int lda, const __bf16* __restrict__ W, int ldw,
    float* __restrict__ C, int ldc, int M, int N, int K) {
  int wid  = blockIdx.x * 8 + (threadIdx.x >> 5);
  int lane = threadIdx.x & 31;
  int ntiles = N >> 6;
  if (wid >= (M >> 6) * ntiles) return;
  int n0 = (wid % ntiles) << 6;
  int m0 = (wid / ntiles) << 6;
  int lm = lane & 15, hi = lane >> 4;

  v8f acc[4][4];
#pragma unroll
  for (int i = 0; i < 4; ++i)
#pragma unroll
    for (int j = 0; j < 4; ++j) acc[i][j] = (v8f){};

  const __bf16* ap[4];
  const __bf16* bp[4];
#pragma unroll
  for (int i = 0; i < 4; ++i) {
    ap[i] = A + (size_t)(m0 + i * 16 + lm) * lda + hi * 16;
    bp[i] = W + (size_t)(n0 + i * 16 + lm) * ldw + hi * 16;
  }
  for (int k0 = 0; k0 < K; k0 += 32) {
    v16bf a[4], b[4];
#pragma unroll
    for (int i = 0; i < 4; ++i) a[i] = load_a_frag(ap[i] + k0);
#pragma unroll
    for (int i = 0; i < 4; ++i) b[i] = load_b_frag(bp[i] + k0);
#pragma unroll
    for (int i = 0; i < 4; ++i)
#pragma unroll
      for (int j = 0; j < 4; ++j) acc[i][j] = wmma_bf16(a[i], b[j], acc[i][j]);
  }
#pragma unroll
  for (int i = 0; i < 4; ++i)
#pragma unroll
    for (int j = 0; j < 4; ++j)
#pragma unroll
      for (int r = 0; r < 8; ++r)   // C layout: row = base + r + 8*hi, col = lm
        C[(size_t)(m0 + i * 16 + r + 8 * hi) * ldc + n0 + j * 16 + lm] = acc[i][j][r];
}

// ---------------------------------------------------- flash causal MQA kernel
// Q:[B,H,L,128] bf16, K:[B,L,128] bf16, Vt:[B,128,L] bf16, O:[B,L,H*128] bf16
// One block (8 waves) = 128 consecutive queries of one (b,h).  K/V tiles are
// staged per 32-key block into double-buffered LDS with async global->LDS
// copies (ASYNCcnt); block i+1's transfers overlap block i's WMMA/softmax.
#define KSTR 136   // 32 x 128 K tile, padded rows (272B, 16B-aligned)
#define VSTR 40    // 128 x 32 V^T tile, padded rows (80B, 16B-aligned)
__global__ void __launch_bounds__(256) mqa_flash_kernel(
    const __bf16* __restrict__ Q, const __bf16* __restrict__ Kr,
    const __bf16* __restrict__ Vt, __bf16* __restrict__ O,
    int B, int H, int L) {
  __shared__ __align__(16) __bf16 ksh[2][32 * KSTR];
  __shared__ __align__(16) __bf16 vsh[2][128 * VSTR];
  __shared__ __align__(16) __bf16 pbuf[8][16 * 48];
  int wslot = threadIdx.x >> 5;
  int lane  = threadIdx.x & 31;
  int btiles = L >> 7;
  int Q0 = (blockIdx.x % btiles) << 7;
  int h  = (blockIdx.x / btiles) % H;
  int b  = blockIdx.x / (btiles * H);
  int q0 = Q0 + (wslot << 4);
  int lm = lane & 15, hi = lane >> 4;

  const __bf16* qrow  = Q + (((size_t)(b * H + h) * L) + q0 + lm) * 128 + hi * 16;
  const __bf16* kbase = Kr + ((size_t)b * L) * 128;
  const __bf16* vbase = Vt + ((size_t)b * 128) * L;

  v16bf qa[4];
#pragma unroll
  for (int c = 0; c < 4; ++c) qa[c] = load_a_frag(qrow + c * 32);

  v8f accO[8];
#pragma unroll
  for (int t = 0; t < 8; ++t) accO[t] = (v8f){};
  float mrow[8], lrow[8];
#pragma unroll
  for (int r = 0; r < 8; ++r) { mrow[r] = -1e30f; lrow[r] = 0.0f; }

  // cooperative staging indices (256 threads move 8KB K + 8KB V per block;
  // each wave issues exactly 4 async instructions per block)
  int skey  = threadIdx.x >> 3;            // 0..31
  int shd16 = (threadIdx.x & 7) << 4;      // 0,16,...,112
  int vhd   = threadIdx.x >> 1;            // 0..127
  int vkh   = (threadIdx.x & 1) << 4;      // 0 / 16
  auto stage_blk = [&](int buf, int k0) {
    stage32(&ksh[buf][skey * KSTR + shd16], kbase + (size_t)(k0 + skey) * 128 + shd16);
    stage32(&vsh[buf][vhd * VSTR + vkh],    vbase + (size_t)vhd * L + k0 + vkh);
  };

  const float scale = 0.08838834764831845f;  // 1/sqrt(128)
  __bf16* pw = &pbuf[wslot][0];
  int kendW = q0 + 15;                       // this wave's last valid key
  int nblk  = ((Q0 + 127) >> 5) + 1;         // key blocks for whole workgroup

  stage_blk(0, 0);                           // prologue: fill buffer 0
  for (int blk = 0; blk < nblk; ++blk) {
    int cur = blk & 1;
    int k0  = blk << 5;
    bool has_next = (blk + 1) < nblk;        // block-uniform
    if (has_next) stage_blk(1 - cur, k0 + 32);
    if (has_next) stage_wait4(); else stage_wait0();
    __syncthreads();  // current tile visible to all waves
    if (k0 <= kendW) {                       // wave-uniform causal skip
      const __bf16* ks = &ksh[cur][0];
      const __bf16* vs = &vsh[cur][0];
      v8f s0 = {}, s1 = {};
#pragma unroll
      for (int c = 0; c < 4; ++c) {          // S = Q * K^T over hd chunks
        v16bf b0 = load_b_frag(ks + lm * KSTR + c * 32 + hi * 16);
        v16bf b1 = load_b_frag(ks + (16 + lm) * KSTR + c * 32 + hi * 16);
        s0 = wmma_bf16(qa[c], b0, s0);
        s1 = wmma_bf16(qa[c], b1, s1);
      }
      float alpha[8];
#pragma unroll
      for (int r = 0; r < 8; ++r) {
        int row  = q0 + r + 8 * hi;
        float v0 = (k0 + lm      <= row) ? s0[r] * scale : -1e30f;
        float v1 = (k0 + 16 + lm <= row) ? s1[r] * scale : -1e30f;
        float mx = fmaxf(v0, v1);
        mx = fmaxf(mx, __shfl_xor(mx, 1, 32));
        mx = fmaxf(mx, __shfl_xor(mx, 2, 32));
        mx = fmaxf(mx, __shfl_xor(mx, 4, 32));
        mx = fmaxf(mx, __shfl_xor(mx, 8, 32));
        float mnew = fmaxf(mrow[r], mx);
        alpha[r] = __expf(mrow[r] - mnew);
        mrow[r]  = mnew;
        float p0 = __expf(v0 - mnew);
        float p1 = __expf(v1 - mnew);
        s0[r] = p0; s1[r] = p1;
        float ps = p0 + p1;
        ps += __shfl_xor(ps, 1, 32);
        ps += __shfl_xor(ps, 2, 32);
        ps += __shfl_xor(ps, 4, 32);
        ps += __shfl_xor(ps, 8, 32);
        lrow[r] = lrow[r] * alpha[r] + ps;
      }
#pragma unroll
      for (int t = 0; t < 8; ++t)
#pragma unroll
        for (int r = 0; r < 8; ++r) accO[t][r] *= alpha[r];
      // C-layout -> A-layout via per-wave LDS round trip
#pragma unroll
      for (int r = 0; r < 8; ++r) {
        pw[(r + 8 * hi) * 48 + lm]      = f32_to_bf16(s0[r]);
        pw[(r + 8 * hi) * 48 + 16 + lm] = f32_to_bf16(s1[r]);
      }
      v16bf pa = load_a_frag(pw + lm * 48 + hi * 16);
#pragma unroll
      for (int t = 0; t < 8; ++t) {          // O += P * V (hd tiles of 16)
        v16bf bv = load_b_frag(vs + (t * 16 + lm) * VSTR + hi * 16);
        accO[t] = wmma_bf16(pa, bv, accO[t]);
      }
    }
    __syncthreads();  // all reads done; next iteration may refill this buffer
  }
#pragma unroll
  for (int r = 0; r < 8; ++r) {
    float inv = 1.0f / lrow[r];
    __bf16* op = O + ((size_t)(b * L + q0 + r + 8 * hi)) * (H * 128) + h * 128;
#pragma unroll
    for (int t = 0; t < 8; ++t) op[t * 16 + lm] = f32_to_bf16(accO[t][r] * inv);
  }
}

// --------------------------------------------------------------------- launch
extern "C" void kernel_launch(void* const* d_in, const int* in_sizes, int n_in,
                              void* d_out, int out_size, void* d_ws, size_t ws_size,
                              hipStream_t stream) {
  const float* x    = (const float*)d_in[0];
  const float* cosT = (const float*)d_in[1];
  const float* sinT = (const float*)d_in[2];
  const float* wq   = (const float*)d_in[3];
  const float* wk   = (const float*)d_in[4];
  const float* wv   = (const float*)d_in[5];
  const float* wo   = (const float*)d_in[6];
  float* out = (float*)d_out;

  const int B = 2, L = 2048, D = 2048, H = 16, HD = 128;
  const int M = B * L;

  char* ws = (char*)d_ws;
  size_t off = 0;
  auto alloc = [&](size_t bytes) {
    char* p = ws + off;
    off += (bytes + 255) & ~(size_t)255;
    return p;
  };
  __bf16* xb  = (__bf16*)alloc((size_t)M * D * 2);
  __bf16* wqb = (__bf16*)alloc((size_t)D * D * 2);
  __bf16* wkb = (__bf16*)alloc((size_t)HD * D * 2);
  __bf16* wvb = (__bf16*)alloc((size_t)HD * D * 2);
  __bf16* wob = (__bf16*)alloc((size_t)D * D * 2);
  float*  qf  = (float*)alloc((size_t)M * D * 4);
  float*  kf  = (float*)alloc((size_t)M * HD * 4);
  float*  vf  = (float*)alloc((size_t)M * HD * 4);
  __bf16* qb  = (__bf16*)alloc((size_t)M * D * 2);
  __bf16* kb  = (__bf16*)alloc((size_t)M * HD * 2);
  __bf16* vtb = (__bf16*)alloc((size_t)M * HD * 2);
  __bf16* ob  = (__bf16*)qf;  // reuse qf region after rope consumes it

  auto cvt = [&](const float* s, __bf16* d, int n) {
    cvt_bf16_kernel<<<(n + 255) / 256, 256, 0, stream>>>(s, d, n);
  };
  cvt(x, xb, M * D);
  cvt(wq, wqb, D * D);
  cvt(wk, wkb, HD * D);
  cvt(wv, wvb, HD * D);
  cvt(wo, wob, D * D);

  auto gemm = [&](const __bf16* A, int lda, const __bf16* W, int ldw, float* C,
                  int ldc, int Mm, int Nn, int Kk) {
    int tiles = (Mm / 64) * (Nn / 64);
    gemm_bf16_nt<<<(tiles + 7) / 8, 256, 0, stream>>>(A, lda, W, ldw, C, ldc, Mm, Nn, Kk);
  };
  gemm(xb, D, wqb, D, qf, D, M, D, D);    // Q projection
  gemm(xb, D, wkb, D, kf, HD, M, HD, D);  // K projection
  gemm(xb, D, wvb, D, vf, HD, M, HD, D);  // V projection

  {
    int waves = B * H * L;
    rope_rms_kernel<<<(waves + 7) / 8, 256, 0, stream>>>(qf, cosT, sinT, qb, B, H, L);
  }
  {
    int waves = B * L;
    rope_rms_kernel<<<(waves + 7) / 8, 256, 0, stream>>>(kf, cosT, sinT, kb, B, 1, L);
  }
  {
    int n = M * HD;
    vtrans_kernel<<<(n + 255) / 256, 256, 0, stream>>>(vf, vtb, B, L);
  }
  {
    int blocks = B * H * (L / 128);
    mqa_flash_kernel<<<blocks, 256, 0, stream>>>(qb, kb, vtb, ob, B, H, L);
  }
  gemm(ob, D, wob, D, out, D, M, D, D);   // output projection

  (void)in_sizes; (void)n_in; (void)out_size; (void)ws_size;
}
